// Convoluzione_70866960384316
// MI455X (gfx1250) — matrix operand order repaired and checked by reading, not compiled, not run on hardware
//
#include <hip/hip_runtime.h>

#define BATCH 32
#define C_IN  128
#define C_OUT 256
#define Hdim  56
#define HO    54
#define NPOS  (HO*HO)                       // 2916
#define NTILE 64
#define NTILES ((NPOS + NTILE - 1)/NTILE)   // 46

typedef __attribute__((ext_vector_type(16))) __bf16 v16bf;
typedef __attribute__((ext_vector_type(8)))  float  v8f;

static constexpr size_t IPART = (size_t)BATCH * Hdim * Hdim * C_IN;  // image elems / part
static constexpr size_t WPART = (size_t)9 * C_OUT * C_IN;            // weight elems / part

__device__ __forceinline__ unsigned short f2bf(float f){
  union { float f; unsigned u; } cv; cv.f = f;
  unsigned u = cv.u;
  unsigned r = u + 0x7fffu + ((u >> 16) & 1u);   // round-to-nearest-even
  return (unsigned short)(r >> 16);
}
__device__ __forceinline__ float bf2f(unsigned short s){
  union { unsigned u; float f; } cv; cv.u = ((unsigned)s) << 16;
  return cv.f;
}

// NCHW f32 image -> [part][n][y][x][ci] bf16 (hi, and residual lo if nparts==2).
// B fragment (lane = N + 16*(K>=16), 16 contiguous K per lane) is then one 32B load.
__global__ void cvt_img_kernel(const float* __restrict__ img,
                               unsigned short* __restrict__ wsI, int nparts){
  size_t i = (size_t)blockIdx.x * blockDim.x + threadIdx.x;
  if (i >= IPART) return;
  int ci = (int)(i & (C_IN - 1));
  size_t r = i >> 7;
  int x = (int)(r % Hdim); r /= Hdim;
  int y = (int)(r % Hdim);
  int n = (int)(r / Hdim);
  float v = img[(((size_t)n * C_IN + ci) * Hdim + y) * Hdim + x];
  unsigned short h = f2bf(v);
  wsI[i] = h;
  if (nparts == 2) {
    wsI[IPART + i] = f2bf(v - bf2f(h));
  }
}

// OIHW f32 filter -> [part][t][co][ci] bf16 (hi + residual lo).
// Within each 32-ci chunk ci keeps its natural order (K = 16*kh + e) so a lane's
// 16 values (K runs 16kh..16kh+15, same as B) are CONTIGUOUS:
//   dest j = c*32 + kh*16 + e  <-  src ci = c*32 + 16*kh + e
__global__ void cvt_w_kernel(const float* __restrict__ flt,
                             unsigned short* __restrict__ wsW, int nparts){
  size_t i = (size_t)blockIdx.x * blockDim.x + threadIdx.x;
  if (i >= WPART) return;
  int j  = (int)(i & (C_IN - 1));
  size_t r = i >> 7;
  int co = (int)(r % C_OUT);
  int t  = (int)(r / C_OUT);                     // t = ky*3+kx
  int c  = j >> 5;
  int jj = j & 31;
  int kh = jj >> 4;
  int e  = jj & 15;
  int K  = 16 * kh + e;
  int ci = c * 32 + K;
  float v = flt[((size_t)co * C_IN + ci) * 9 + t];
  unsigned short h = f2bf(v);
  wsW[i] = h;
  if (nparts == 2) {
    wsW[WPART + i] = f2bf(v - bf2f(h));
  }
}

// Implicit-GEMM conv: per block -> all 256 co x 64 positions of one batch image.
// 8 waves; wave w owns co rows [32w, 32w+32). K loop: 9 taps x 4 chunks of 32 ci.
// bf16x3 split (hi*hi + hi*lo + lo*hi) recovers ~fp32 accuracy when NP==2.
template<int NP>
__global__ __launch_bounds__(256)
void conv_wmma_kernel(const unsigned short* __restrict__ wsI_,
                      const unsigned short* __restrict__ wsW_,
                      float* __restrict__ out){
  const __bf16* wsI = (const __bf16*)wsI_;
  const __bf16* wsW = (const __bf16*)wsW_;
  const int n    = blockIdx.y;
  const int p0   = blockIdx.x * NTILE;
  const int tid  = threadIdx.x;
  const int wave = tid >> 5;
  const int lane = tid & 31;
  const int ml   = lane & 15;    // M row / N col within 16
  const int kh   = lane >> 4;    // K-half selector
  const int co0  = wave * 32;

  // Per-lane output positions for the 4 N-subtiles (clamped; masked at store).
  int py[4], px[4];
#pragma unroll
  for (int nt = 0; nt < 4; ++nt) {
    int p = p0 + nt * 16 + ml;
    if (p > NPOS - 1) p = NPOS - 1;
    py[nt] = p / HO;
    px[nt] = p - py[nt] * HO;
  }

  v8f acc[2][4];
#pragma unroll
  for (int mi = 0; mi < 2; ++mi)
#pragma unroll
    for (int nt = 0; nt < 4; ++nt)
#pragma unroll
      for (int r = 0; r < 8; ++r) acc[mi][nt][r] = 0.0f;

  for (int t = 0; t < 9; ++t) {
    const int ky = t / 3;
    const int kx = t - ky * 3;

    // Per-tap base pointers (fragment-contiguous thanks to ws layouts).
    const __bf16* bBase[4][2];
#pragma unroll
    for (int nt = 0; nt < 4; ++nt) {
      const size_t off =
          (((size_t)n * Hdim + (py[nt] + ky)) * Hdim + (px[nt] + kx)) * C_IN + kh * 16;
#pragma unroll
      for (int pp = 0; pp < NP; ++pp) bBase[nt][pp] = wsI + (size_t)pp * IPART + off;
    }
    const __bf16* aBase[2][2];
#pragma unroll
    for (int mi = 0; mi < 2; ++mi) {
      const size_t off =
          ((size_t)t * C_OUT + (co0 + mi * 16 + ml)) * C_IN + kh * 16;
#pragma unroll
      for (int pp = 0; pp < NP; ++pp) aBase[mi][pp] = wsW + (size_t)pp * WPART + off;
    }

#pragma unroll
    for (int c = 0; c < 4; ++c) {
      const int ci0 = c * 32;

      v16bf A[2][2];
#pragma unroll
      for (int mi = 0; mi < 2; ++mi)
#pragma unroll
        for (int pp = 0; pp < NP; ++pp)
          A[mi][pp] = *(const v16bf*)(aBase[mi][pp] + ci0);

      v16bf Bf[4][2];
#pragma unroll
      for (int nt = 0; nt < 4; ++nt)
#pragma unroll
        for (int pp = 0; pp < NP; ++pp)
          Bf[nt][pp] = *(const v16bf*)(bBase[nt][pp] + ci0);

#pragma unroll
      for (int mi = 0; mi < 2; ++mi)
#pragma unroll
        for (int nt = 0; nt < 4; ++nt) {
          acc[mi][nt] = __builtin_amdgcn_wmma_f32_16x16x32_bf16(
              false, A[mi][0], false, Bf[nt][0], (short)0, acc[mi][nt], false, false);
          if (NP == 2) {
            acc[mi][nt] = __builtin_amdgcn_wmma_f32_16x16x32_bf16(
                false, A[mi][0], false, Bf[nt][1], (short)0, acc[mi][nt], false, false);
            acc[mi][nt] = __builtin_amdgcn_wmma_f32_16x16x32_bf16(
                false, A[mi][1], false, Bf[nt][0], (short)0, acc[mi][nt], false, false);
          }
        }
    }
  }

  // C/D layout: lane<16 -> M=r, N=lane ; lane>=16 -> M=8+r, N=lane-16.
#pragma unroll
  for (int mi = 0; mi < 2; ++mi) {
    const int coBase = co0 + mi * 16 + kh * 8;
#pragma unroll
    for (int nt = 0; nt < 4; ++nt) {
      const int p = p0 + nt * 16 + ml;
      if (p < NPOS) {
#pragma unroll
        for (int r = 0; r < 8; ++r)
          out[((size_t)n * C_OUT + (coBase + r)) * NPOS + p] = acc[mi][nt][r];
      }
    }
  }
}

// Fallback if workspace is too small for bf16 staging.
__global__ void conv_naive_kernel(const float* __restrict__ img,
                                  const float* __restrict__ flt,
                                  float* __restrict__ out){
  size_t i = (size_t)blockIdx.x * blockDim.x + threadIdx.x;
  size_t total = (size_t)BATCH * C_OUT * NPOS;
  if (i >= total) return;
  int p  = (int)(i % NPOS);
  int co = (int)((i / NPOS) % C_OUT);
  int n  = (int)(i / ((size_t)NPOS * C_OUT));
  int y = p / HO, x = p - (p / HO) * HO;
  float s = 0.f;
  for (int ci = 0; ci < C_IN; ++ci)
    for (int ky = 0; ky < 3; ++ky)
      for (int kx = 0; kx < 3; ++kx)
        s += img[(((size_t)n * C_IN + ci) * Hdim + (y + ky)) * Hdim + (x + kx)] *
             flt[(((size_t)co * C_IN + ci) * 3 + ky) * 3 + kx];
  out[i] = s;
}

extern "C" void kernel_launch(void* const* d_in, const int* in_sizes, int n_in,
                              void* d_out, int out_size, void* d_ws, size_t ws_size,
                              hipStream_t stream) {
  (void)in_sizes; (void)n_in; (void)out_size;
  const float* img = (const float*)d_in[0];
  const float* flt = (const float*)d_in[1];
  float* out = (float*)d_out;

  const size_t need1 = (IPART + WPART) * sizeof(unsigned short);   // ~26.3 MB
  const int np = (ws_size >= 2 * need1) ? 2 : (ws_size >= need1 ? 1 : 0);

  if (np == 0) {
    size_t total = (size_t)BATCH * C_OUT * NPOS;
    conv_naive_kernel<<<(unsigned)((total + 255) / 256), 256, 0, stream>>>(img, flt, out);
    return;
  }

  unsigned short* wsI = (unsigned short*)d_ws;
  unsigned short* wsW = wsI + (size_t)np * IPART;

  cvt_img_kernel<<<(unsigned)((IPART + 255) / 256), 256, 0, stream>>>(img, wsI, np);
  cvt_w_kernel  <<<(unsigned)((WPART + 255) / 256), 256, 0, stream>>>(flt, wsW, np);

  dim3 grid(NTILES, BATCH);
  if (np == 2) conv_wmma_kernel<2><<<grid, 256, 0, stream>>>(wsI, wsW, out);
  else         conv_wmma_kernel<1><<<grid, 256, 0, stream>>>(wsI, wsW, out);
}
